// SparseMoEFFN_28484223107689
// MI455X (gfx1250) — matrix operand order repaired and checked by reading, not compiled, not run on hardware
//
#include <hip/hip_runtime.h>
#include <hip/hip_bf16.h>
#include <math.h>

// ---------------- problem constants ----------------
#define TOK   32768      // B*S tokens
#define DIM   512        // model dim D
#define FDIM  2048       // ffn dim F
#define NEXP  8          // experts
#define TOPK  2

typedef __attribute__((ext_vector_type(16))) __bf16 v16bf;
typedef __attribute__((ext_vector_type(8)))  __bf16 v8bf;
typedef __attribute__((ext_vector_type(4)))  __bf16 v4bf;
typedef __attribute__((ext_vector_type(8)))  float  v8f;

union V16 { v16bf v; v8bf h[2]; };

// ---------------- workspace layout (bytes) ----------------
#define OFF_W1T   ((size_t)0)                                   // [E][F][D] bf16 : 16 MB
#define OFF_W2T   (OFF_W1T + (size_t)NEXP*FDIM*DIM*2)           // [E][D][F] bf16 : 16 MB
#define OFF_XN    (OFF_W2T + (size_t)NEXP*FDIM*DIM*2)           // [T][D]   bf16 : 32 MB
#define OFF_YS    (OFF_XN  + (size_t)TOK*DIM*2)                 // [T*K][D] bf16 : 64 MB
#define OFF_SW    (OFF_YS  + (size_t)TOK*TOPK*DIM*2)            // [T*K]    f32  : 256 KB
#define OFF_LIST  (OFF_SW  + (size_t)TOK*TOPK*4)                // [E][T]   int  : 1 MB
#define OFF_CNT   (OFF_LIST+ (size_t)NEXP*TOK*4)                // [E]      int

// ============================================================
// 1) weight conversion + transpose, fp32 -> bf16
//    w1 [E,D,F] -> w1t [E,F,D]   (B-operand for GEMM1: K=D contiguous)
//    w2 [E,F,D] -> w2t [E,D,F]   (B-operand for GEMM2: K=F contiguous)
// ============================================================
__global__ __launch_bounds__(256) void cvt_w1_kernel(const float* __restrict__ w1,
                                                     __bf16* __restrict__ w1t) {
    size_t i = (size_t)blockIdx.x * 256 + threadIdx.x;   // over [E][D][F]
    int f = (int)(i % FDIM);
    size_t r = i / FDIM;
    int d = (int)(r % DIM);
    int e = (int)(r / DIM);
    w1t[((size_t)e * FDIM + f) * DIM + d] = (__bf16)w1[i];
}

__global__ __launch_bounds__(256) void cvt_w2_kernel(const float* __restrict__ w2,
                                                     __bf16* __restrict__ w2t) {
    size_t i = (size_t)blockIdx.x * 256 + threadIdx.x;   // over [E][F][D]
    int d = (int)(i % DIM);
    size_t r = i / DIM;
    int f = (int)(r % FDIM);
    int e = (int)(r / FDIM);
    w2t[((size_t)e * DIM + d) * FDIM + f] = (__bf16)w2[i];
}

// ============================================================
// 2) router: layernorm -> bf16 xn, logits, softmax, top-2,
//    append slot ids to per-expert lists (wave32, 1 token/wave)
// ============================================================
__global__ __launch_bounds__(256) void router_kernel(const float* __restrict__ x,
                                                     const float* __restrict__ gate_w,
                                                     const float* __restrict__ ln_g,
                                                     const float* __restrict__ ln_b,
                                                     __bf16* __restrict__ xn,
                                                     float*  __restrict__ slotw,
                                                     int*    __restrict__ counts,
                                                     int*    __restrict__ lists) {
    const int lane = threadIdx.x & 31;
    const int wv   = threadIdx.x >> 5;
    const int t    = blockIdx.x * 8 + wv;
    const int d0   = lane * 16;

    const float* xp = x + (size_t)t * DIM + d0;
    float v[16];
    float s = 0.f, ss = 0.f;
#pragma unroll
    for (int i = 0; i < 16; ++i) { v[i] = xp[i]; s += v[i]; ss += v[i] * v[i]; }
#pragma unroll
    for (int off = 16; off > 0; off >>= 1) {
        s  += __shfl_xor(s,  off, 32);
        ss += __shfl_xor(ss, off, 32);
    }
    const float mu   = s  * (1.f / DIM);
    const float var  = ss * (1.f / DIM) - mu * mu;
    const float rstd = rsqrtf(var + 1e-5f);

    float lacc[NEXP];
#pragma unroll
    for (int e = 0; e < NEXP; ++e) lacc[e] = 0.f;

#pragma unroll
    for (int i = 0; i < 16; ++i) {
        float n = (v[i] - mu) * rstd * ln_g[d0 + i] + ln_b[d0 + i];
        v[i] = n;
        const float* gw = gate_w + (size_t)(d0 + i) * NEXP;
#pragma unroll
        for (int e = 0; e < NEXP; ++e) lacc[e] += n * gw[e];
    }

    // store bf16 x_norm (two 16B stores per lane)
    v8bf lo, hi;
#pragma unroll
    for (int i = 0; i < 8; ++i) { lo[i] = (__bf16)v[i]; hi[i] = (__bf16)v[i + 8]; }
    *(v8bf*)&xn[(size_t)t * DIM + d0]     = lo;
    *(v8bf*)&xn[(size_t)t * DIM + d0 + 8] = hi;

#pragma unroll
    for (int off = 16; off > 0; off >>= 1)
#pragma unroll
        for (int e = 0; e < NEXP; ++e) lacc[e] += __shfl_xor(lacc[e], off, 32);

    if (lane == 0) {
        float m = lacc[0];
#pragma unroll
        for (int e = 1; e < NEXP; ++e) m = fmaxf(m, lacc[e]);
        float p[NEXP]; float sum = 0.f;
#pragma unroll
        for (int e = 0; e < NEXP; ++e) { p[e] = __expf(lacc[e] - m); sum += p[e]; }
        const float inv = 1.f / sum;
        int i1 = 0; float b1v = p[0];
#pragma unroll
        for (int e = 1; e < NEXP; ++e) if (p[e] > b1v) { b1v = p[e]; i1 = e; }
        int i2 = (i1 == 0) ? 1 : 0; float b2v = p[i2];
#pragma unroll
        for (int e = 0; e < NEXP; ++e)
            if (e != i1 && p[e] > b2v) { b2v = p[e]; i2 = e; }

        const int   ei[2] = { i1, i2 };
        const float wi[2] = { b1v * inv, b2v * inv };
#pragma unroll
        for (int k = 0; k < TOPK; ++k) {
            int slot = t * TOPK + k;
            slotw[slot] = wi[k];
            int pos = atomicAdd(&counts[ei[k]], 1);
            lists[(size_t)ei[k] * TOK + pos] = slot;
        }
    }
}

// ============================================================
// 3) pad each expert list to a multiple of 16 with -1
// ============================================================
__global__ void pad_kernel(const int* __restrict__ counts, int* __restrict__ lists) {
    int e = blockIdx.x;
    int c = counts[e];
    int padded = (c + 15) & ~15;
    int idx = c + threadIdx.x;
    if (idx < padded) lists[(size_t)e * TOK + idx] = -1;
}

// ============================================================
// 4) fused expert FFN: Y = GELU(X*W1 + b1)*W2 + b2, WMMA bf16
//    block = 4 waves, one 16-row token tile per block
// ============================================================
__global__ __launch_bounds__(128) void ffn_kernel(const __bf16* __restrict__ xn,
                                                  const __bf16* __restrict__ w1t,
                                                  const __bf16* __restrict__ w2t,
                                                  const float*  __restrict__ b1,
                                                  const float*  __restrict__ b2,
                                                  const int*    __restrict__ counts,
                                                  const int*    __restrict__ lists,
                                                  __bf16*       __restrict__ yslots) {
    const int e    = blockIdx.y;
    const int tile = blockIdx.x;
    const int cnt  = counts[e];
    const int padded = (cnt + 15) & ~15;
    if (tile * 16 >= padded) return;

    __shared__ __bf16 Xa[16][DIM + 8];   // token-tile A operand, +16B row pad
    __shared__ __bf16 Hs[16][64 + 8];    // GELU'd hidden chunk (A for GEMM2)
    __shared__ int    Ssl[16];

    const int tid  = threadIdx.x;
    const int lane = tid & 31;
    const int wv   = tid >> 5;
    const int nsel = lane & 15;              // N (and A-row M) selector
    const int halfA = (lane < 16) ? 0 : 8;   // A K-offset per lane half
    const int halfB = (lane < 16) ? 0 : 16;  // unused; B loads use halfA (same k order as A)
    const int mbase = (lane < 16) ? 0 : 8;   // C/D row base per lane half

    if (tid < 16) Ssl[tid] = lists[(size_t)e * TOK + tile * 16 + tid];
    __syncthreads();

    // gather 16 token rows of bf16 x_norm into LDS (16B chunks)
    for (int c = tid; c < 16 * (DIM / 8); c += 128) {
        int m  = c >> 6;
        int k8 = (c & 63) << 3;
        int slot = Ssl[m];
        int tok  = (slot >= 0) ? (slot >> 1) : 0;
        *(v8bf*)&Xa[m][k8] = *(const v8bf*)&xn[(size_t)tok * DIM + k8];
    }
    __syncthreads();

    v8f yacc[8];
#pragma unroll
    for (int j = 0; j < 8; ++j)
#pragma unroll
        for (int r = 0; r < 8; ++r) yacc[j][r] = 0.f;

    for (int fc = 0; fc < FDIM / 64; ++fc) {
        // ---- stage 1: H[16x16 per wave] = GELU(X @ W1[:,chunk] + b1) ----
        const int fcol = fc * 64 + wv * 16 + nsel;
        const __bf16* w1p = w1t + ((size_t)e * FDIM + fcol) * DIM;
        v8f hacc;
#pragma unroll
        for (int r = 0; r < 8; ++r) hacc[r] = 0.f;

#pragma unroll
        for (int ks = 0; ks < DIM / 32; ++ks) {
            V16 a, b;
            a.h[0] = *(const v8bf*)&Xa[nsel][ks * 32 + halfA];
            a.h[1] = *(const v8bf*)&Xa[nsel][ks * 32 + 16 + halfA];
            b.h[0] = *(const v8bf*)&w1p[ks * 32 + halfA];
            b.h[1] = *(const v8bf*)&w1p[ks * 32 + 16 + halfA];
            hacc = __builtin_amdgcn_wmma_f32_16x16x32_bf16(
                false, a.v, false, b.v, (short)0, hacc, false, false);
        }

        const float bb1 = b1[e * FDIM + fcol];
#pragma unroll
        for (int r = 0; r < 8; ++r) {
            float h = hacc[r] + bb1;
            float g = 0.5f * h * (1.0f + erff(h * 0.70710678118f));  // exact GELU
            Hs[mbase + r][wv * 16 + nsel] = (__bf16)g;
        }
        __syncthreads();

        // ---- stage 2: Y[16x512] += H[16x64] @ W2[chunk,:] ----
#pragma unroll
        for (int j = 0; j < 8; ++j) {
            const int n0 = (wv * 8 + j) * 16;
            const __bf16* w2p =
                w2t + ((size_t)e * DIM + n0 + nsel) * FDIM + fc * 64;
#pragma unroll
            for (int kk = 0; kk < 2; ++kk) {
                V16 a, b;
                a.h[0] = *(const v8bf*)&Hs[nsel][kk * 32 + halfA];
                a.h[1] = *(const v8bf*)&Hs[nsel][kk * 32 + 16 + halfA];
                b.h[0] = *(const v8bf*)&w2p[kk * 32 + halfA];
                b.h[1] = *(const v8bf*)&w2p[kk * 32 + 16 + halfA];
                yacc[j] = __builtin_amdgcn_wmma_f32_16x16x32_bf16(
                    false, a.v, false, b.v, (short)0, yacc[j], false, false);
            }
        }
        __syncthreads();
    }

    // ---- write Y tile to per-slot buffer (bf16) ----
#pragma unroll
    for (int j = 0; j < 8; ++j) {
        const int n0 = (wv * 8 + j) * 16;
        const float bb2 = b2[e * DIM + n0 + nsel];
#pragma unroll
        for (int r = 0; r < 8; ++r) {
            int slot = Ssl[mbase + r];
            if (slot >= 0)
                yslots[(size_t)slot * DIM + n0 + nsel] = (__bf16)(yacc[j][r] + bb2);
        }
    }
}

// ============================================================
// 5) combine: out = x + w0*y(slot0) + w1*y(slot1)
// ============================================================
__global__ __launch_bounds__(128) void combine_kernel(const float*  __restrict__ x,
                                                      const float*  __restrict__ slotw,
                                                      const __bf16* __restrict__ yslots,
                                                      float* __restrict__ out) {
    const int t  = blockIdx.x;
    const int d  = threadIdx.x * 4;
    const float w0 = slotw[t * TOPK + 0];
    const float w1 = slotw[t * TOPK + 1];
    const float4 xv = *(const float4*)&x[(size_t)t * DIM + d];
    v4bf y0 = *(const v4bf*)&yslots[((size_t)t * TOPK + 0) * DIM + d];
    v4bf y1 = *(const v4bf*)&yslots[((size_t)t * TOPK + 1) * DIM + d];
    float4 o;
    o.x = xv.x + w0 * (float)y0[0] + w1 * (float)y1[0];
    o.y = xv.y + w0 * (float)y0[1] + w1 * (float)y1[1];
    o.z = xv.z + w0 * (float)y0[2] + w1 * (float)y1[2];
    o.w = xv.w + w0 * (float)y0[3] + w1 * (float)y1[3];
    *(float4*)&out[(size_t)t * DIM + d] = o;
}

// ============================================================
extern "C" void kernel_launch(void* const* d_in, const int* in_sizes, int n_in,
                              void* d_out, int out_size, void* d_ws, size_t ws_size,
                              hipStream_t stream) {
    const float* x      = (const float*)d_in[0];
    const float* gate_w = (const float*)d_in[1];
    const float* w1     = (const float*)d_in[2];
    const float* b1     = (const float*)d_in[3];
    const float* w2     = (const float*)d_in[4];
    const float* b2     = (const float*)d_in[5];
    const float* ln_g   = (const float*)d_in[6];
    const float* ln_b   = (const float*)d_in[7];
    float* out = (float*)d_out;

    char* ws = (char*)d_ws;
    __bf16* w1t    = (__bf16*)(ws + OFF_W1T);
    __bf16* w2t    = (__bf16*)(ws + OFF_W2T);
    __bf16* xn     = (__bf16*)(ws + OFF_XN);
    __bf16* yslots = (__bf16*)(ws + OFF_YS);
    float*  slotw  = (float*) (ws + OFF_SW);
    int*    lists  = (int*)   (ws + OFF_LIST);
    int*    counts = (int*)   (ws + OFF_CNT);

    // zero expert counters (memset node is graph-capturable)
    hipMemsetAsync(counts, 0, NEXP * sizeof(int), stream);

    // 1) weight convert + transpose (fp32 -> bf16)
    {
        const size_t n = (size_t)NEXP * DIM * FDIM;
        const int blocks = (int)(n / 256);
        cvt_w1_kernel<<<blocks, 256, 0, stream>>>(w1, w1t);
        cvt_w2_kernel<<<blocks, 256, 0, stream>>>(w2, w2t);
    }

    // 2) router: LN + softmax + top-2 + list build (8 tokens / block)
    router_kernel<<<TOK / 8, 256, 0, stream>>>(x, gate_w, ln_g, ln_b,
                                               xn, slotw, counts, lists);

    // 3) pad lists to 16-row tiles
    pad_kernel<<<NEXP, 16, 0, stream>>>(counts, lists);

    // 4) fused expert FFN (WMMA bf16). Max tiles per expert = TOK/16.
    {
        dim3 grid(TOK / 16, NEXP);
        ffn_kernel<<<grid, 128, 0, stream>>>(xn, w1t, w2t, b1, b2,
                                             counts, lists, yslots);
    }

    // 5) residual combine
    combine_kernel<<<TOK, DIM / 4, 0, stream>>>(x, slotw, yslots, out);
}
